// fbp_39487929319487
// MI455X (gfx1250) — matrix operand order repaired and checked by reading, not compiled, not run on hardware
//
#include <hip/hip_runtime.h>
#include <hip/hip_bf16.h>

// ---------------------------------------------------------------------------
// FBP: ramp filter as WMMA GEMM (f16 in / f32 acc) + bilinear backprojection.
//   A = 360 angles, D = 320 detectors, I = 320 image, batch B = 2.
//   filtered[m][d] = sum_k x[m][k] * g(d-k),  m = b*360+a
//   g(0)=0.5, g(t)=-2/(pi t)^2 for odd t, 0 for even t  (exact spatial form
//   of the reference's FFT ramp filter; zero-pad to 1024 >= 2D-1 => linear)
// ---------------------------------------------------------------------------

#define N_ANGLES   360
#define DET_COUNT  320
#define IMG        320
#define M_ROWS     (2 * N_ANGLES)          // 720
#define PIX_PER_B  (IMG * IMG)             // 102400
#define SINO_PER_B (N_ANGLES * DET_COUNT)  // 115200

static constexpr double kPI = 3.14159265358979323846;

typedef __attribute__((ext_vector_type(16))) _Float16 v16h;
typedef __attribute__((ext_vector_type(8)))  _Float16 v8h;
typedef __attribute__((ext_vector_type(8)))  float    v8f;

// ---------------------------------------------------------------------------
// Prep 1: angle tables — cos(theta_a), sin(theta_a), ix_a = (xang_a+1)*179.5
// thetas = linspace(0, pi - pi/A, A)  => theta_a = a*pi/360
// xang   = linspace(-1, 1, A)
// ---------------------------------------------------------------------------
__global__ void fbp_tables_kernel(float* __restrict__ tabs) {
    int a = blockIdx.x * blockDim.x + threadIdx.x;
    if (a >= N_ANGLES) return;
    double th = (double)a * kPI / 360.0;
    tabs[a]              = (float)cos(th);
    tabs[N_ANGLES + a]   = (float)sin(th);
    float xang = (float)(-1.0 + 2.0 * (double)a / (double)(N_ANGLES - 1));
    // reference: ix = (gx + 1.0) * (W - 1) * 0.5 in f32
    tabs[2 * N_ANGLES + a] = ((xang + 1.0f) * (float)(N_ANGLES - 1)) * 0.5f;
}

// ---------------------------------------------------------------------------
// Prep 2: Gt (N x K, row-major), f16.  Gt[n][k] = g(n - k).
// ---------------------------------------------------------------------------
__global__ void fbp_gt_kernel(_Float16* __restrict__ Gt) {
    int idx = blockIdx.x * blockDim.x + threadIdx.x;
    if (idx >= DET_COUNT * DET_COUNT) return;
    int n = idx / DET_COUNT;
    int k = idx % DET_COUNT;
    int t = n - k;
    double v;
    if (t == 0)            v = 0.5;
    else if (t & 1)        v = -2.0 / (kPI * kPI * (double)t * (double)t);
    else                   v = 0.0;
    Gt[idx] = (_Float16)v;
}

// ---------------------------------------------------------------------------
// Prep 3: X f32 -> f16 copy (720 x 320 row-major, same flat layout as input).
// ---------------------------------------------------------------------------
__global__ void fbp_x16_kernel(const float* __restrict__ x,
                               _Float16* __restrict__ X16) {
    int idx = blockIdx.x * blockDim.x + threadIdx.x;
    if (idx >= M_ROWS * DET_COUNT) return;
    X16[idx] = (_Float16)x[idx];
}

// ---------------------------------------------------------------------------
// Filter GEMM: C[720x320] = X16[720x320(K)] * Gt16[320(N)x320(K)]^T
// One wave per 16x16 output tile; K-loop in steps of 32 via
// v_wmma_f32_16x16x32_f16.  Fragment layouts per CDNA5 ISA 7.12.2:
//   A (16x32 f16): lane l: M = l&15; halves 0..7  -> K = k0 + (l>>4)*8 + t
//                                   halves 8..15 -> K = k0 + 16 + (l>>4)*8 + t
//   B (32x16 f16): lane l: N = l&15; same K mapping as A
//   C/D (16x16 f32): vgpr r: lanes 0-15 -> M = r, lanes 16-31 -> M = r+8
// ---------------------------------------------------------------------------
__global__ void fbp_filter_wmma_kernel(const _Float16* __restrict__ X16,
                                       const _Float16* __restrict__ Gt16,
                                       float* __restrict__ C) {
    const int NT = DET_COUNT / 16;                       // 20 tiles in N
    const int MT = M_ROWS / 16;                          // 45 tiles in M
    int wave = (blockIdx.x * blockDim.x + threadIdx.x) >> 5;
    int lane = threadIdx.x & 31;
    int tm = wave / NT;
    int tn = wave % NT;
    if (tm >= MT) return;                                // whole-wave exit

    int l15 = lane & 15;
    int khi = lane >> 4;                                 // 0 or 1
    int m0  = tm * 16;
    int n0  = tn * 16;

    const _Float16* arow = X16  + (size_t)(m0 + l15) * DET_COUNT;
    const _Float16* brow = Gt16 + (size_t)(n0 + l15) * DET_COUNT;

    v8f acc = {};
    for (int k0 = 0; k0 < DET_COUNT; k0 += 32) {
        const _Float16* ap = arow + k0 + khi * 8;
        v8h alo = *(const v8h*)(ap);
        v8h ahi = *(const v8h*)(ap + 16);
        v16h a = __builtin_shufflevector(alo, ahi,
                 0,1,2,3,4,5,6,7,8,9,10,11,12,13,14,15);

        const _Float16* bp = brow + k0 + khi * 8;
        v8h blo = *(const v8h*)(bp);
        v8h bhi = *(const v8h*)(bp + 16);
        v16h b = __builtin_shufflevector(blo, bhi,
                 0,1,2,3,4,5,6,7,8,9,10,11,12,13,14,15);

        acc = __builtin_amdgcn_wmma_f32_16x16x32_f16(
                  /*neg_a=*/false, a, /*neg_b=*/false, b,
                  /*c_mod=*/(short)0, acc,
                  /*reuse_a=*/false, /*reuse_b=*/false);
    }

    int cn    = n0 + l15;
    int mbase = m0 + khi * 8;
#pragma unroll
    for (int r = 0; r < 8; ++r)
        C[(size_t)(mbase + r) * DET_COUNT + cn] = acc[r];
}

// ---------------------------------------------------------------------------
// Backprojection: one thread per output pixel per batch.
// img(y=d, x=a) = C[(b*360 + a)*320 + d]; tables staged in LDS.
// ---------------------------------------------------------------------------
__global__ void fbp_backproject_kernel(const float* __restrict__ C,
                                       const float* __restrict__ tabs,
                                       float* __restrict__ out) {
    __shared__ float sc[N_ANGLES];
    __shared__ float ss[N_ANGLES];
    __shared__ float sx[N_ANGLES];
    for (int t = threadIdx.x; t < N_ANGLES; t += blockDim.x) {
        sc[t] = tabs[t];
        ss[t] = tabs[N_ANGLES + t];
        sx[t] = tabs[2 * N_ANGLES + t];
    }
    __syncthreads();

    int idx = blockIdx.x * blockDim.x + threadIdx.x;
    if (idx >= 2 * PIX_PER_B) return;
    int b = idx / PIX_PER_B;
    int p = idx % PIX_PER_B;
    int i = p / IMG;                    // row -> grid_y = lin[i]
    int j = p % IMG;                    // col -> grid_x = lin[j]

    float lin_i = (float)(-1.0 + 2.0 * (double)i / (double)(IMG - 1));
    float lin_j = (float)(-1.0 + 2.0 * (double)j / (double)(IMG - 1));

    const float* img = C + (size_t)b * SINO_PER_B;

    float acc = 0.0f;
    for (int a = 0; a < N_ANGLES; ++a) {
        float t  = lin_j * sc[a] - lin_i * ss[a];
        float iy = ((t + 1.0f) * (float)(DET_COUNT - 1)) * 0.5f;
        float ix = sx[a];

        float x0 = floorf(ix), y0 = floorf(iy);
        float x1 = x0 + 1.0f,  y1 = y0 + 1.0f;
        float wa = (x1 - ix) * (y1 - iy);
        float wb = (ix - x0) * (y1 - iy);
        float wc = (x1 - ix) * (iy - y0);
        float wd = (ix - x0) * (iy - y0);

        auto samp = [&](float yf, float xf) -> float {
            bool valid = (xf >= 0.0f) & (xf <= (float)(N_ANGLES - 1)) &
                         (yf >= 0.0f) & (yf <= (float)(DET_COUNT - 1));
            int xc = (int)fminf(fmaxf(xf, 0.0f), (float)(N_ANGLES - 1));
            int yc = (int)fminf(fmaxf(yf, 0.0f), (float)(DET_COUNT - 1));
            return valid ? img[xc * DET_COUNT + yc] : 0.0f;
        };

        acc += wa * samp(y0, x0) + wb * samp(y0, x1) +
               wc * samp(y1, x0) + wd * samp(y1, x1);
    }

    // step_size = I/D = 1 ; scale = pi / (2*A)
    out[idx] = acc * (float)(kPI / (2.0 * (double)N_ANGLES));
}

// ---------------------------------------------------------------------------
// Launch
// ---------------------------------------------------------------------------
extern "C" void kernel_launch(void* const* d_in, const int* in_sizes, int n_in,
                              void* d_out, int out_size, void* d_ws, size_t ws_size,
                              hipStream_t stream) {
    const float* x = (const float*)d_in[0];      // (2,1,360,320) f32
    float* out = (float*)d_out;                  // (2,1,320,320) f32

    char* ws = (char*)d_ws;
    float*     Cbuf = (float*)ws;                              // 921600 B
    _Float16*  X16  = (_Float16*)(ws + 921600);                // 460800 B
    _Float16*  Gt16 = (_Float16*)(ws + 1382400);               // 204800 B
    float*     tabs = (float*)(ws + 1587200);                  //   4320 B

    fbp_tables_kernel<<<(N_ANGLES + 255) / 256, 256, 0, stream>>>(tabs);
    fbp_gt_kernel<<<(DET_COUNT * DET_COUNT + 255) / 256, 256, 0, stream>>>(Gt16);
    fbp_x16_kernel<<<(M_ROWS * DET_COUNT + 255) / 256, 256, 0, stream>>>(x, X16);

    // 45 * 20 = 900 waves; 8 waves (256 threads) per block
    int nwaves = (M_ROWS / 16) * (DET_COUNT / 16);
    int nblocks = (nwaves + 7) / 8;
    fbp_filter_wmma_kernel<<<nblocks, 256, 0, stream>>>(X16, Gt16, Cbuf);

    fbp_backproject_kernel<<<(2 * PIX_PER_B + 255) / 256, 256, 0, stream>>>(
        Cbuf, tabs, out);
}
